// xSiamese_74414603370677
// MI455X (gfx1250) — matrix operand order repaired and checked by reading, not compiled, not run on hardware
//
#include <hip/hip_runtime.h>
#include <hip/hip_bf16.h>

typedef __bf16 bf16_t;
typedef __attribute__((ext_vector_type(16))) __bf16 v16bf;
typedef __attribute__((ext_vector_type(8)))  float  v8f;
typedef __attribute__((ext_vector_type(4)))  int    i32x4;
typedef __attribute__((ext_vector_type(4)))  unsigned u32x4;
typedef __attribute__((ext_vector_type(8)))  unsigned u32x8;

union Frag { i32x4 q[2]; v16bf v; };
static_assert(sizeof(Frag) == 32, "frag size");

// ---------------- workspace layout (bytes) ----------------
static const size_t WDS_OFF   = 0;                          // 256*1024 bf16
static const size_t WK_OFF    = WDS_OFF   + 524288;         // 256*2304 bf16
static const size_t WS_OFF    = WK_OFF    + 1179648;        // 256*2304 bf16
static const size_t WF_OFF    = WS_OFF    + 1179648;        // 256*256  bf16
static const size_t REFT_OFF  = WF_OFF    + 131072;         // 64*49*1024 bf16 (cropped ref, NHWC)
static const size_t A1_OFF    = REFT_OFF  + 6422528;        // 64*49*256 bf16
static const size_t KB_OFF    = A1_OFF    + 1605632;        // 64*25*256 bf16
static const size_t SRCHT_OFF = KB_OFF    + 819200;         // 64*961*1024 bf16 (search, NHWC)
static const size_t A2_OFF    = SRCHT_OFF + 125960192;      // 64*961*256 bf16
static const size_t SB_OFF    = SRCHT_OFF;                  // reuse: 64*841*256 bf16
static const size_t CORR_OFF  = A2_OFF;                     // reuse: 64*625*256 bf16

// ---------------- weight prep: fold BN gamma, convert to bf16 ----------------
__global__ void prep_w1x1(const float* __restrict__ w, const float* __restrict__ g,
                          bf16_t* __restrict__ out, int cin, int n) {
  int i = blockIdx.x * blockDim.x + threadIdx.x;
  if (i < n) out[i] = (bf16_t)(w[i] * g[i / cin]);
}

// OIHW (256,256,3,3) -> [oc][(dy*3+dx)*256 + ci], gamma-scaled
__global__ void prep_w3x3(const float* __restrict__ w, const float* __restrict__ g,
                          bf16_t* __restrict__ out, int n) {
  int i = blockIdx.x * blockDim.x + threadIdx.x;
  if (i >= n) return;
  int oc  = i / (9 * 256);
  int r   = i % (9 * 256);
  int tap = r >> 8;
  int ci  = r & 255;
  out[i] = (bf16_t)(w[(oc * 256 + ci) * 9 + tap] * g[oc]);
}

// ---------------- NCHW f32 -> NHWC bf16, with spatial crop ----------------
__global__ void nchw_to_nhwc_bf16(const float* __restrict__ in, bf16_t* __restrict__ out,
                                  int C, int H, int W, int OH, int OW, int y0, int x0) {
  __shared__ float tile[32][33];
  const int npos = OH * OW;
  const int b  = blockIdx.z;
  const int c0 = blockIdx.y * 32;
  const int p0 = blockIdx.x * 32;
  const int tx = threadIdx.x & 31;
  const int tg = threadIdx.x >> 5;  // 0..7
  #pragma unroll
  for (int i = 0; i < 4; ++i) {
    int cl = tg + i * 8;
    int p  = p0 + tx;
    int pc = p < npos ? p : npos - 1;
    int y = pc / OW, x = pc % OW;
    tile[cl][tx] = in[(((size_t)b * C + c0 + cl) * H + y0 + y) * W + x0 + x];
  }
  __syncthreads();
  #pragma unroll
  for (int i = 0; i < 4; ++i) {
    int pl = tg + i * 8;
    int p  = p0 + pl;
    if (p < npos)
      out[((size_t)b * npos + p) * C + c0 + tx] = (bf16_t)tile[tx][pl];
  }
}

// ---------------- generic conv (1x1 / 3x3, VALID) as bf16 WMMA GEMM ----------------
// in : NHWC bf16 [B][H][W][Cin]
// wgt: [oc][KH*KW*Cin] bf16 (gamma folded), bias: f32 [256] (BN beta)
// out: NHWC bf16 [B][OH*OW][256], or f32 NCHW [B][256][OH*OW] (nchw_f32)
// wave tile: 32 positions x 64 oc (8 WMMA / K-chunk, B-fragments reused across the
// two position fragments); block = 8 waves = 256 positions x 64 oc
__global__ void conv_wmma(const bf16_t* __restrict__ in, const bf16_t* __restrict__ wgt,
                          const float* __restrict__ bias,
                          bf16_t* __restrict__ outb, float* __restrict__ outf,
                          int H, int W, int OH, int OW, int y0, int x0,
                          int KH, int KW, int Cin, int relu, int nchw_f32) {
  const int lane = threadIdx.x & 31;
  const int wv   = threadIdx.x >> 5;
  const int half = lane >> 4;   // 0 | 1
  const int nlo  = lane & 15;
  const int npos = OH * OW;
  const int p0   = blockIdx.x * 256 + wv * 32;
  const int oc0  = blockIdx.y * 64;
  const int b    = blockIdx.z;
  const int KK   = KH * KW * Cin;

  // two A-fragment rows (output positions) for this lane, clamped (stores guarded)
  int oyv[2], oxv[2];
  #pragma unroll
  for (int u = 0; u < 2; ++u) {
    int p  = p0 + u * 16 + nlo;
    int pc = p < npos ? p : npos - 1;
    oyv[u] = pc / OW;
    oxv[u] = pc % OW;
  }

  const bf16_t* inb = in + (size_t)b * H * W * Cin;

  v8f acc[2][4];
  #pragma unroll
  for (int j = 0; j < 4; ++j) {
    float bv = bias[oc0 + j * 16 + nlo];  // D: N = lane&15 -> per-lane bias broadcast
    #pragma unroll
    for (int u = 0; u < 2; ++u)
      #pragma unroll
      for (int e = 0; e < 8; ++e) acc[u][j][e] = bv;
  }

  for (int t = 0; t < KH * KW; ++t) {
    int dy = t / KW, dx = t % KW;
    const bf16_t* arow0 = inb + ((size_t)(y0 + oyv[0] + dy) * W + (x0 + oxv[0] + dx)) * Cin;
    const bf16_t* arow1 = inb + ((size_t)(y0 + oyv[1] + dy) * W + (x0 + oxv[1] + dx)) * Cin;
    for (int c0 = 0; c0 < Cin; c0 += 32) {
      // A fragment 16x32: lane half 0 holds K {0..7,16..23}, half 1 holds {8..15,24..31}
      Frag a0, a1;
      const bf16_t* ap0 = arow0 + c0 + half * 8;
      const bf16_t* ap1 = arow1 + c0 + half * 8;
      a0.q[0] = *(const i32x4*)ap0;
      a0.q[1] = *(const i32x4*)(ap0 + 16);
      a1.q[0] = *(const i32x4*)ap1;
      a1.q[1] = *(const i32x4*)(ap1 + 16);
      __builtin_prefetch(arow0 + c0 + 64, 0, 1);  // global_prefetch_b8
      __builtin_prefetch(arow1 + c0 + 64, 0, 1);
      const int kb = t * Cin + c0 + half * 8;     // B: K {0..7,16..23} / {8..15,24..31} at this lane's column
      #pragma unroll
      for (int j = 0; j < 4; ++j) {
        Frag bf;
        const bf16_t* bp = wgt + (size_t)(oc0 + j * 16 + nlo) * KK + kb;
        bf.q[0] = *(const i32x4*)bp;
        bf.q[1] = *(const i32x4*)(bp + 16);
        acc[0][j] = __builtin_amdgcn_wmma_f32_16x16x32_bf16(
            false, a0.v, false, bf.v, (short)0, acc[0][j], false, false);
        acc[1][j] = __builtin_amdgcn_wmma_f32_16x16x32_bf16(
            false, a1.v, false, bf.v, (short)0, acc[1][j], false, false);
      }
    }
  }

  // D layout: VGPR r holds M = r + 8*half, N = lane&15
  #pragma unroll
  for (int u = 0; u < 2; ++u) {
    #pragma unroll
    for (int j = 0; j < 4; ++j) {
      int oc = oc0 + j * 16 + nlo;
      #pragma unroll
      for (int r = 0; r < 8; ++r) {
        int pp = p0 + u * 16 + half * 8 + r;
        if (pp < npos) {
          float v = acc[u][j][r];
          if (relu) v = fmaxf(v, 0.0f);
          if (nchw_f32)
            outf[((size_t)b * 256 + oc) * npos + pp] = v;
          else
            outb[((size_t)b * npos + pp) * 256 + oc] = (bf16_t)v;
        }
      }
    }
  }
}

// ---------------- depthwise cross-correlation (5x5 over 29x29 -> 25x25) + ReLU ----------------
// S: [B][29][29][256] bf16, K: [B][5][5][256] bf16, out: [B][25][25][256] bf16
// block = (b, oy). The 5 search rows needed (full width, NHWC) are one contiguous
// 74240 B stretch; the 5x5 kernel tile is one contiguous 12800 B stretch. Both are
// staged into dynamic LDS by the Tensor Data Mover (degenerate 2-D descriptor:
// tile_dim0 = N elements, tile_dim1 = 1, data_size = 2 B), then reduced on VALU.
#define SS_ELEMS (5 * 29 * 256)   // 37120
#define SK_ELEMS (25 * 256)       // 6400

#if defined(__gfx1250__)
__device__ inline void tdm_load_1d(unsigned lds_byte_off, const bf16_t* gptr, unsigned nelem) {
  unsigned long long ga = (unsigned long long)(size_t)gptr;
  u32x4 g0;
  g0[0] = 1u;                                   // count=1, user descriptor
  g0[1] = lds_byte_off;                         // lds_addr
  g0[2] = (unsigned)ga;                         // global_addr[31:0]
  g0[3] = (unsigned)((ga >> 32) & 0x01FFFFFFu)  // global_addr[56:32]
          | 0x80000000u;                        // type = 2 ("image")
  u32x8 g1;
  g1[0] = 1u << 16;                             // workgroup_mask=0, data_size=1 (2 bytes)
  g1[1] = (nelem & 0xFFFFu) << 16;              // tensor_dim0[15:0]
  g1[2] = ((nelem >> 16) & 0xFFFFu)             // tensor_dim0[31:16]
          | (1u << 16);                         // tensor_dim1 = 1 (low 16)
  g1[3] = (nelem & 0xFFFFu) << 16;              // tensor_dim1 hi=0 | tile_dim0 = nelem
  g1[4] = 1u;                                   // tile_dim1 = 1, tile_dim2 = 0
  g1[5] = nelem;                                // tensor_dim0_stride[31:0]
  g1[6] = 0u;                                   // stride0 hi | stride1 lo
  g1[7] = 0u;
  asm volatile("tensor_load_to_lds %0, %1" :: "s"(g0), "s"(g1) : "memory");
}
#endif

__global__ void dwcorr(const bf16_t* __restrict__ S, const bf16_t* __restrict__ K,
                       bf16_t* __restrict__ out) {
  extern __shared__ __align__(16) char smem[];
  bf16_t* sS = (bf16_t*)smem;
  bf16_t* sK = (bf16_t*)(smem + SS_ELEMS * 2);
  const int bi = blockIdx.x;
  const int b = bi / 25, oy = bi % 25;
  const int tid = threadIdx.x;

  const bf16_t* sbase = S + ((size_t)b * 841 + oy * 29) * 256;  // rows oy..oy+4, contiguous
  const bf16_t* kbase = K + (size_t)b * SK_ELEMS;

#if defined(__gfx1250__)
  if (tid < 32) {  // wave 0 drives the TDM; EXEC-independent DMA, scalar descriptors
    unsigned lds0 = __builtin_amdgcn_groupstaticsize();
    tdm_load_1d(lds0, sbase, SS_ELEMS);
    tdm_load_1d(lds0 + SS_ELEMS * 2, kbase, SK_ELEMS);
    __builtin_amdgcn_s_wait_tensorcnt(0);
  }
#else
  for (int u = tid; u < SS_ELEMS / 8; u += 256)
    *(i32x4*)(sS + u * 8) = *(const i32x4*)(sbase + u * 8);
  for (int u = tid; u < SK_ELEMS / 8; u += 256)
    *(i32x4*)(sK + u * 8) = *(const i32x4*)(kbase + u * 8);
#endif
  __syncthreads();

  for (int e = tid; e < 25 * 256; e += 256) {
    int ox = e >> 8, c = e & 255;
    float acc = 0.0f;
    #pragma unroll
    for (int dy = 0; dy < 5; ++dy)
      #pragma unroll
      for (int dx = 0; dx < 5; ++dx)
        acc += (float)sS[(dy * 29 + ox + dx) * 256 + c] * (float)sK[(dy * 5 + dx) * 256 + c];
    out[((size_t)b * 625 + oy * 25 + ox) * 256 + c] = (bf16_t)fmaxf(acc, 0.0f);
  }
}

extern "C" void kernel_launch(void* const* d_in, const int* in_sizes, int n_in,
                              void* d_out, int out_size, void* d_ws, size_t ws_size,
                              hipStream_t stream) {
  (void)in_sizes; (void)n_in; (void)out_size; (void)ws_size;
  const float* ref  = (const float*)d_in[0];
  const float* srch = (const float*)d_in[1];
  const float* w_ds = (const float*)d_in[2];
  const float* g_ds = (const float*)d_in[3];
  const float* b_ds = (const float*)d_in[4];
  const float* w_k  = (const float*)d_in[5];
  const float* g_k  = (const float*)d_in[6];
  const float* b_k  = (const float*)d_in[7];
  const float* w_s  = (const float*)d_in[8];
  const float* g_s  = (const float*)d_in[9];
  const float* b_s  = (const float*)d_in[10];
  const float* w_f  = (const float*)d_in[11];
  const float* g_f  = (const float*)d_in[12];
  const float* b_f  = (const float*)d_in[13];
  float* out = (float*)d_out;
  char*  ws  = (char*)d_ws;

  bf16_t* WDS   = (bf16_t*)(ws + WDS_OFF);
  bf16_t* WK    = (bf16_t*)(ws + WK_OFF);
  bf16_t* WS    = (bf16_t*)(ws + WS_OFF);
  bf16_t* WF    = (bf16_t*)(ws + WF_OFF);
  bf16_t* REFT  = (bf16_t*)(ws + REFT_OFF);
  bf16_t* A1    = (bf16_t*)(ws + A1_OFF);
  bf16_t* KB    = (bf16_t*)(ws + KB_OFF);
  bf16_t* SRCHT = (bf16_t*)(ws + SRCHT_OFF);
  bf16_t* A2    = (bf16_t*)(ws + A2_OFF);
  bf16_t* SBp   = (bf16_t*)(ws + SB_OFF);
  bf16_t* CORR  = (bf16_t*)(ws + CORR_OFF);

  // 1) fold BN gamma into bf16 weights
  prep_w1x1<<<(256 * 1024 + 255) / 256, 256, 0, stream>>>(w_ds, g_ds, WDS, 1024, 256 * 1024);
  prep_w3x3<<<(256 * 2304 + 255) / 256, 256, 0, stream>>>(w_k, g_k, WK, 256 * 2304);
  prep_w3x3<<<(256 * 2304 + 255) / 256, 256, 0, stream>>>(w_s, g_s, WS, 256 * 2304);
  prep_w1x1<<<(256 * 256 + 255) / 256, 256, 0, stream>>>(w_f, g_f, WF, 256, 256 * 256);

  // 2) NCHW f32 -> NHWC bf16 (ref: only the 7x7 center survives the crop)
  nchw_to_nhwc_bf16<<<dim3(2, 32, 64), 256, 0, stream>>>(ref,  REFT,  1024, 15, 15, 7, 7, 4, 4);
  nchw_to_nhwc_bf16<<<dim3(31, 32, 64), 256, 0, stream>>>(srch, SRCHT, 1024, 31, 31, 31, 31, 0, 0);

  // 3) downsample0: 1x1 conv (1024->256) + BN
  conv_wmma<<<dim3(1, 4, 64), 256, 0, stream>>>(REFT,  WDS, b_ds, A1, nullptr,
                                                7, 7, 7, 7, 0, 0, 1, 1, 1024, 0, 0);
  conv_wmma<<<dim3(4, 4, 64), 256, 0, stream>>>(SRCHT, WDS, b_ds, A2, nullptr,
                                                31, 31, 31, 31, 0, 0, 1, 1, 1024, 0, 0);

  // 4) 3x3 conv branches + BN + ReLU
  conv_wmma<<<dim3(1, 4, 64), 256, 0, stream>>>(A1, WK, b_k, KB, nullptr,
                                                7, 7, 5, 5, 0, 0, 3, 3, 256, 1, 0);
  conv_wmma<<<dim3(4, 4, 64), 256, 0, stream>>>(A2, WS, b_s, SBp, nullptr,
                                                31, 31, 29, 29, 0, 0, 3, 3, 256, 1, 0);

  // 5) depthwise cross-correlation + ReLU (TDM-staged LDS)
  dwcorr<<<dim3(64 * 25), 256, (SS_ELEMS + SK_ELEMS) * 2, stream>>>(SBp, KB, CORR);

  // 6) fuse: 1x1 conv + BN + ReLU -> f32 NCHW into d_out
  conv_wmma<<<dim3(3, 4, 64), 256, 0, stream>>>(CORR, WF, b_f, nullptr, out,
                                                25, 25, 25, 25, 0, 0, 1, 1, 256, 1, 1);
}
